// LSTM_5454608466030
// MI455X (gfx1250) — matrix operand order repaired and checked by reading, not compiled, not run on hardware
//
#include <hip/hip_runtime.h>

// Problem sizes (match reference)
constexpr int kB = 128;   // batch
constexpr int kT = 256;   // timesteps
constexpr int kD = 512;   // input dim
constexpr int kH = 1024;  // hidden dim
constexpr int kC = 1024;  // classes
constexpr int kNWG = 64;  // persistent workgroups: one 16-col stripe of H each
constexpr int kThr = 256; // 8 wave32 per WG: one 16-row tile of B each

// LDS layout: [4][16][kH] recurrent weights (128 KB) + [4][16][kD] input weights (64 KB)
constexpr int kLdsWhElems = 4 * 16 * kH;
constexpr int kLdsWxElems = 4 * 16 * kD;
constexpr int kLdsBytes   = (kLdsWhElems + kLdsWxElems) * 2;  // 192 KB (<320 KB/WG)

typedef __attribute__((ext_vector_type(16))) __bf16 v16bf;
typedef __attribute__((ext_vector_type(8)))  float  v8f;

union Frag {
  v16bf v;
  uint4 q[2];
};

__device__ __forceinline__ unsigned short f2bf(float f) {
  union { float f; unsigned u; } v; v.f = f;
  unsigned r = v.u + 0x7FFFu + ((v.u >> 16) & 1u);  // round-to-nearest-even
  return (unsigned short)(r >> 16);
}

__device__ __forceinline__ v8f wmma_bf16(const Frag& a, const Frag& b, v8f c) {
  return __builtin_amdgcn_wmma_f32_16x16x32_bf16(
      /*neg_a=*/false, a.v, /*neg_b=*/false, b.v,
      /*c_mod=*/(short)0, c, /*reuse_a=*/false, /*reuse_b=*/false);
}

__device__ __forceinline__ float sigmoid_fast(float x) {
  return 1.0f / (1.0f + __expf(-x));  // v_exp_f32 + v_rcp_f32
}

__device__ __forceinline__ float tanh_fast(float x) {
  float e = __expf(-2.0f * x);        // single v_exp_f32
  return (1.0f - e) / (1.0f + e);
}

// ---------------- prep kernels ----------------

__global__ void cvt_x_bf16(const float* __restrict__ src,
                           unsigned short* __restrict__ dst, int n) {
  int i = blockIdx.x * blockDim.x + threadIdx.x;
  int stride = gridDim.x * blockDim.x;
  for (; i < n; i += stride) dst[i] = f2bf(src[i]);
}

// src fp32 [K][N] row-major  ->  dst bf16 [N][K] (column-major of original)
__global__ void transpose_cvt(const float* __restrict__ src,
                              unsigned short* __restrict__ dst, int K, int N) {
  int i = blockIdx.x * blockDim.x + threadIdx.x;
  int total = K * N;
  int stride = gridDim.x * blockDim.x;
  for (; i < total; i += stride) {
    int n = i / K;
    int k = i - n * K;
    dst[i] = f2bf(src[k * N + n]);
  }
}

__global__ void init_state(unsigned short* __restrict__ h0,
                           unsigned short* __restrict__ h1,
                           unsigned* __restrict__ bar, int n) {
  int i = blockIdx.x * blockDim.x + threadIdx.x;
  if (i == 0) { bar[0] = 0u; bar[1] = 0u; }
  int stride = gridDim.x * blockDim.x;
  for (; i < n; i += stride) { h0[i] = 0; h1[i] = 0; }
}

// ---------------- grid-wide barrier (sense via monotonically increasing step) ----------------

__device__ __forceinline__ void grid_barrier(unsigned* __restrict__ bar, int step) {
  __syncthreads();
  if (threadIdx.x == 0) {
    __threadfence();
    unsigned prev = atomicAdd(&bar[0], 1u);
    if (prev == (unsigned)(kNWG - 1)) {
      bar[0] = 0u;
      __threadfence();
      atomicExch(&bar[1], (unsigned)(step + 1));
    } else {
      while (atomicAdd(&bar[1], 0u) < (unsigned)(step + 1)) {
        __builtin_amdgcn_s_sleep(8);
      }
    }
    __threadfence();
  }
  __syncthreads();
}

// ---------------- persistent LSTM kernel ----------------
// Global layouts (all bf16):
//   xbf : [B][T][D]   (x converted)
//   WxT : [4][H][D]   gate-major, column-major of W*x  (WxT[g][n][k] = W[k][n])
//   WhT : [4][H][H]   column-major of W*h
//   WpT : [C][H]      column-major of Wph
//   h buffers: [B][H], double buffered
__global__ __launch_bounds__(kThr)
void lstm_persistent(const unsigned short* __restrict__ xbf,
                     const unsigned short* __restrict__ WxT,
                     const unsigned short* __restrict__ WhT,
                     const unsigned short* __restrict__ WpT,
                     const float* __restrict__ b_i, const float* __restrict__ b_f,
                     const float* __restrict__ b_g, const float* __restrict__ b_o,
                     const float* __restrict__ b_p,
                     unsigned short* __restrict__ h0buf,
                     unsigned short* __restrict__ h1buf,
                     unsigned* __restrict__ bar,
                     float* __restrict__ out) {
  extern __shared__ unsigned short lds[];
  unsigned short* ldsWh = lds;                 // [4][16][kH]
  unsigned short* ldsWx = lds + kLdsWhElems;   // [4][16][kD]

  const int wg   = blockIdx.x;        // column-stripe id: output cols n0..n0+15
  const int n0   = wg * 16;
  const int wave = threadIdx.x >> 5;  // row-tile id: batch rows m0..m0+15
  const int lane = threadIdx.x & 31;
  const int m0   = wave * 16;
  const int lrow = lane & 15;
  const bool hi  = lane >= 16;

  // Stage this stripe's weights into LDS once (reused for all 256 steps).
  // Per gate, source regions WhT[g][n0..n0+15][*] / WxT[g][n0..n0+15][*] are contiguous.
  for (int g = 0; g < 4; ++g) {
    {
      const uint4* src = (const uint4*)(WhT + (size_t)g * kH * kH + (size_t)n0 * kH);
      uint4* dst = (uint4*)(ldsWh + (size_t)g * 16 * kH);
      const int nvec = (16 * kH) / 8;  // 8 bf16 per uint4
      for (int i = threadIdx.x; i < nvec; i += kThr) dst[i] = src[i];
    }
    {
      const uint4* src = (const uint4*)(WxT + (size_t)g * kH * kD + (size_t)n0 * kD);
      uint4* dst = (uint4*)(ldsWx + (size_t)g * 16 * kD);
      const int nvec = (16 * kD) / 8;
      for (int i = threadIdx.x; i < nvec; i += kThr) dst[i] = src[i];
    }
  }
  __syncthreads();

  const int col = n0 + lrow;  // this lane's output column (C/D layout: N = lane%16)
  const float bias0 = b_i[col], bias1 = b_f[col], bias2 = b_g[col], bias3 = b_o[col];

  const int kb_a = hi ? 8 : 0;   // A fragment K sub-offsets (ISA A layout)
  const int kb_b = hi ? 8 : 0;   // B fragment K sub-offsets (same order as A)

  v8f c_state = {};  // cell state tile lives in VGPRs for the whole sequence

  const unsigned short* hbuf[2] = { h0buf, h1buf };

  for (int t = 0; t < kT; ++t) {
    const unsigned short* hcur = hbuf[t & 1];
    unsigned short* hnxt = (unsigned short*)hbuf[(t + 1) & 1];

    v8f acc0 = {}, acc1 = {}, acc2 = {}, acc3 = {};

    // ---- x contribution: K = kD, weights from LDS ----
    {
      const unsigned short* xrow =
          xbf + ((size_t)(m0 + lrow) * kT + (size_t)t) * kD;
      const unsigned short* lw = ldsWx + (size_t)lrow * kD + kb_b;
#pragma unroll 4
      for (int ks = 0; ks < kD; ks += 32) {
        Frag a;
        a.q[0] = *(const uint4*)(xrow + ks + kb_a);
        a.q[1] = *(const uint4*)(xrow + ks + kb_a + 16);
        Frag fb;
        const unsigned short* p0 = lw + ks;
        fb.q[0] = *(const uint4*)(p0);
        fb.q[1] = *(const uint4*)(p0 + 16);
        acc0 = wmma_bf16(a, fb, acc0);
        const unsigned short* p1 = p0 + 1 * 16 * kD;
        fb.q[0] = *(const uint4*)(p1);
        fb.q[1] = *(const uint4*)(p1 + 16);
        acc1 = wmma_bf16(a, fb, acc1);
        const unsigned short* p2 = p0 + 2 * 16 * kD;
        fb.q[0] = *(const uint4*)(p2);
        fb.q[1] = *(const uint4*)(p2 + 16);
        acc2 = wmma_bf16(a, fb, acc2);
        const unsigned short* p3 = p0 + 3 * 16 * kD;
        fb.q[0] = *(const uint4*)(p3);
        fb.q[1] = *(const uint4*)(p3 + 16);
        acc3 = wmma_bf16(a, fb, acc3);
      }
    }

    // ---- h contribution: K = kH, weights from LDS ----
    {
      const unsigned short* hrow = hcur + (size_t)(m0 + lrow) * kH;
      const unsigned short* lw = ldsWh + (size_t)lrow * kH + kb_b;
#pragma unroll 4
      for (int ks = 0; ks < kH; ks += 32) {
        Frag a;
        a.q[0] = *(const uint4*)(hrow + ks + kb_a);
        a.q[1] = *(const uint4*)(hrow + ks + kb_a + 16);
        Frag fb;
        const unsigned short* p0 = lw + ks;
        fb.q[0] = *(const uint4*)(p0);
        fb.q[1] = *(const uint4*)(p0 + 16);
        acc0 = wmma_bf16(a, fb, acc0);
        const unsigned short* p1 = p0 + 1 * 16 * kH;
        fb.q[0] = *(const uint4*)(p1);
        fb.q[1] = *(const uint4*)(p1 + 16);
        acc1 = wmma_bf16(a, fb, acc1);
        const unsigned short* p2 = p0 + 2 * 16 * kH;
        fb.q[0] = *(const uint4*)(p2);
        fb.q[1] = *(const uint4*)(p2 + 16);
        acc2 = wmma_bf16(a, fb, acc2);
        const unsigned short* p3 = p0 + 3 * 16 * kH;
        fb.q[0] = *(const uint4*)(p3);
        fb.q[1] = *(const uint4*)(p3 + 16);
        acc3 = wmma_bf16(a, fb, acc3);
      }
    }

    // ---- elementwise LSTM update on the 16x16 tile ----
    v8f hval;
#pragma unroll
    for (int j = 0; j < 8; ++j) {
      float ig = sigmoid_fast(acc0[j] + bias0);
      float fg = sigmoid_fast(acc1[j] + bias1);
      float gg = tanh_fast(acc2[j] + bias2);
      float og = sigmoid_fast(acc3[j] + bias3);
      float cn = gg * ig + c_state[j] * fg;
      c_state[j] = cn;
      hval[j] = tanh_fast(cn) * og;
    }

    // store h tile as bf16 (C/D layout: VGPR j -> M = j or j+8; N = lane%16)
#pragma unroll
    for (int j = 0; j < 8; ++j) {
      int row = m0 + j + (hi ? 8 : 0);
      hnxt[(size_t)row * kH + col] = f2bf(hval[j]);
    }

    grid_barrier(bar, t);
  }

  // ---- final projection p = h @ Wph + bp, fp32 out [B][C] ----
  {
    const unsigned short* hfin = hbuf[kT & 1];
    const unsigned short* hrow = hfin + (size_t)(m0 + lrow) * kH;
    const unsigned short* wprow = WpT + (size_t)(n0 + lrow) * kH;
    v8f accp = {};
#pragma unroll 4
    for (int ks = 0; ks < kH; ks += 32) {
      Frag a;
      a.q[0] = *(const uint4*)(hrow + ks + kb_a);
      a.q[1] = *(const uint4*)(hrow + ks + kb_a + 16);
      Frag fb;
      const unsigned short* p0 = wprow + ks + kb_b;
      fb.q[0] = *(const uint4*)(p0);
      fb.q[1] = *(const uint4*)(p0 + 16);
      accp = wmma_bf16(a, fb, accp);
    }
    const float bpv = b_p[col];
#pragma unroll
    for (int j = 0; j < 8; ++j) {
      int row = m0 + j + (hi ? 8 : 0);
      out[(size_t)row * kC + col] = accp[j] + bpv;
    }
  }
}

// ---------------- host entry ----------------

extern "C" void kernel_launch(void* const* d_in, const int* in_sizes, int n_in,
                              void* d_out, int out_size, void* d_ws, size_t ws_size,
                              hipStream_t stream) {
  (void)in_sizes; (void)n_in; (void)out_size; (void)ws_size;

  const float* x = (const float*)d_in[0];
  const float* Wx[4]   = { (const float*)d_in[1], (const float*)d_in[4],
                           (const float*)d_in[7], (const float*)d_in[10] };
  const float* Wh[4]   = { (const float*)d_in[2], (const float*)d_in[5],
                           (const float*)d_in[8], (const float*)d_in[11] };
  const float* bias[4] = { (const float*)d_in[3], (const float*)d_in[6],
                           (const float*)d_in[9], (const float*)d_in[12] };
  const float* Wp = (const float*)d_in[13];
  const float* bp = (const float*)d_in[14];

  char* ws = (char*)d_ws;
  size_t off = 0;
  auto carve = [&](size_t bytes) -> void* {
    void* p = ws + off;
    off = (off + bytes + 255) & ~(size_t)255;
    return p;
  };
  unsigned short* xbf = (unsigned short*)carve((size_t)kB * kT * kD * 2); // 32 MB
  unsigned short* WxT = (unsigned short*)carve((size_t)4 * kH * kD * 2);  //  4 MB
  unsigned short* WhT = (unsigned short*)carve((size_t)4 * kH * kH * 2);  //  8 MB
  unsigned short* WpT = (unsigned short*)carve((size_t)kC * kH * 2);      //  2 MB
  unsigned short* h0  = (unsigned short*)carve((size_t)kB * kH * 2);
  unsigned short* h1  = (unsigned short*)carve((size_t)kB * kH * 2);
  unsigned* bar       = (unsigned*)carve(256);

  // Prep: bf16 conversions + transposed weight layouts + state init.
  cvt_x_bf16<<<2048, 256, 0, stream>>>(x, xbf, kB * kT * kD);
  for (int g = 0; g < 4; ++g) {
    transpose_cvt<<<1024, 256, 0, stream>>>(Wx[g], WxT + (size_t)g * kH * kD, kD, kH);
    transpose_cvt<<<1024, 256, 0, stream>>>(Wh[g], WhT + (size_t)g * kH * kH, kH, kH);
  }
  transpose_cvt<<<1024, 256, 0, stream>>>(Wp, WpT, kH, kC);
  init_state<<<256, 256, 0, stream>>>(h0, h1, bar, kB * kH);

  // Persistent recurrent kernel: 64 WGs x 256 threads, 192 KB dynamic LDS.
  lstm_persistent<<<kNWG, kThr, kLdsBytes, stream>>>(
      xbf, WxT, WhT, WpT,
      bias[0], bias[1], bias[2], bias[3], bp,
      h0, h1, bar, (float*)d_out);
}
